// FinalSparseTransformerBlock_55284819034474
// MI455X (gfx1250) — compile-verified, matrix operand order checked by reading, not run on hardware
//
#include <hip/hip_runtime.h>

// ---------------------------------------------------------------------------
// Shapes (compile-time constants from the reference)
// ---------------------------------------------------------------------------
#define BQ    4
#define TT    1024
#define DDIM  512
#define HH    8
#define HDIM  4096      // H * D
#define NTOK  4096      // B * T
#define EE    8
#define FF    1024
#define BHH   32        // B * H
#define HALF  256       // D / 2

// GEMM tiling: 8 waves/block (wave32). Block tile 128(M) x 64(N), K-step 32.
// Waves arranged 4(M) x 2(N); each wave owns a 32x32 tile: 2 A fragments x
// 2 B fragments -> 4 WMMAs with both-operand reuse (8 ds_load_b128 / 4 wmma).
// LDS is double-buffered: global loads for step k+1 are issued before the
// WMMAs for step k, one barrier per K-step.
#define BM 128
#define BN 64
#define BK 32

typedef __attribute__((ext_vector_type(16))) __bf16 v16bf;
typedef __attribute__((ext_vector_type(8)))  float  v8f;

// ---------------------------------------------------------------------------
// LDS tiles are stored in *fragment order*: for a K-run of 32, the 16 bf16
// values a lane-half needs (K = kh*8+0..7 and K = 16+kh*8+0..7, kh = lane>>4)
// are contiguous: tile[row][kh][j], j = (k&7) | ((k&16)>>1), kh = (k>>3)&1.
// A fragment load is then one aligned 32-byte LDS read (2x ds_load_b128).
// ---------------------------------------------------------------------------

// ---------------------------------------------------------------------------
// Generic batched bf16-WMMA GEMM, fp32 accumulate. C[z] = A[z] @ B[z].
// TRB: B accessed transposed (B[n*ldb + k]).  Batch offset per z decomposed
// as (z/zdiv, z%zdiv) with dual strides (covers [b,h] attention batching).
// EPI: 0 = store; 1 = C += gvec[row*gstride]*acc (MoE gate-accumulate);
//      2 = causal: C = (col<=row) ? acc*scale : -inf (attention scores).
// ---------------------------------------------------------------------------
template <int TRB, int EPI>
__global__ void __launch_bounds__(256)
gemm_wmma_k(const float* __restrict__ A, const float* __restrict__ B,
            float* __restrict__ C,
            int M, int N, int K, int lda, int ldb, int ldc,
            int zdiv,
            long long sA1, long long sA2, long long sB1, long long sB2,
            long long sC1, long long sC2,
            const float* __restrict__ gvec, int gstride, float scale)
{
  const int z  = blockIdx.z;
  const int zq = z / zdiv, zr = z % zdiv;
  A += zq * sA1 + zr * sA2;
  B += zq * sB1 + zr * sB2;
  C += zq * sC1 + zr * sC2;

  const int m0   = blockIdx.x * BM;
  const int n0   = blockIdx.y * BN;
  const int tid  = threadIdx.x;
  const int lane = tid & 31;
  const int wid  = tid >> 5;
  const int wm   = (wid >> 1) * 32;   // 0/32/64/96
  const int wn   = (wid & 1) * 32;    // 0/32
  const int kh   = lane >> 4;         // lane half (K striping)
  const int lm   = lane & 15;

  __shared__ alignas(32) __bf16 Asf[2][BM * BK];   // 2 x 8 KB, fragment order
  __shared__ alignas(32) __bf16 Bsf[2][BN * BK];   // 2 x 4 KB, fragment order

  v8f acc[2][2];
#pragma unroll
  for (int mt = 0; mt < 2; ++mt)
#pragma unroll
    for (int nt = 0; nt < 2; ++nt)
      acc[mt][nt] = (v8f){0.f,0.f,0.f,0.f,0.f,0.f,0.f,0.f};

  float4 ra[4];    // staged A (128x32 / 256 threads = 4 float4 each)
  float4 rb[2];    // staged B (32x64  / 256 threads = 2 float4 each)

  auto gloadA = [&](int k0) {
#pragma unroll
    for (int i = 0; i < 4; ++i) {
      int linear = (tid + i * 256) << 2;
      int r = linear >> 5, c = linear & 31;
      ra[i] = *(const float4*)(A + (long long)(m0 + r) * lda + (k0 + c));
    }
  };
  auto gloadB = [&](int k0) {
#pragma unroll
    for (int i = 0; i < 2; ++i) {
      int linear = (tid + i * 256) << 2;
      if (TRB) {
        int n = linear >> 5, c = linear & 31;
        rb[i] = *(const float4*)(B + (long long)(n0 + n) * ldb + (k0 + c));
      } else {
        int kk = linear >> 6, nn = linear & 63;
        rb[i] = *(const float4*)(B + (long long)(k0 + kk) * ldb + (n0 + nn));
      }
    }
  };
  auto storeA = [&](int buf) {
#pragma unroll
    for (int i = 0; i < 4; ++i) {
      int linear = (tid + i * 256) << 2;
      int r = linear >> 5, c = linear & 31;
      int ckh = (c >> 3) & 1;
      int cj  = (c & 7) | ((c & 16) >> 1);          // 4 consecutive j's
      __bf16* d = &Asf[buf][r * BK + ckh * 16 + cj];
      d[0] = (__bf16)ra[i].x; d[1] = (__bf16)ra[i].y;
      d[2] = (__bf16)ra[i].z; d[3] = (__bf16)ra[i].w;
    }
  };
  auto storeB = [&](int buf) {
#pragma unroll
    for (int i = 0; i < 2; ++i) {
      int linear = (tid + i * 256) << 2;
      if (TRB) {
        int n = linear >> 5, c = linear & 31;
        int ckh = (c >> 3) & 1;
        int cj  = (c & 7) | ((c & 16) >> 1);
        __bf16* d = &Bsf[buf][n * BK + ckh * 16 + cj];
        d[0] = (__bf16)rb[i].x; d[1] = (__bf16)rb[i].y;
        d[2] = (__bf16)rb[i].z; d[3] = (__bf16)rb[i].w;
      } else {
        int kk = linear >> 6, nn = linear & 63;
        int ckh = (kk >> 3) & 1;
        int cj  = (kk & 7) | ((kk & 16) >> 1);
        Bsf[buf][(nn + 0) * BK + ckh * 16 + cj] = (__bf16)rb[i].x;
        Bsf[buf][(nn + 1) * BK + ckh * 16 + cj] = (__bf16)rb[i].y;
        Bsf[buf][(nn + 2) * BK + ckh * 16 + cj] = (__bf16)rb[i].z;
        Bsf[buf][(nn + 3) * BK + ckh * 16 + cj] = (__bf16)rb[i].w;
      }
    }
  };
  auto compute = [&](int buf) {
    v16bf a0 = *(const v16bf*)&Asf[buf][(wm +      lm) * BK + kh * 16];
    v16bf a1 = *(const v16bf*)&Asf[buf][(wm + 16 + lm) * BK + kh * 16];
    v16bf b0 = *(const v16bf*)&Bsf[buf][(wn +      lm) * BK + kh * 16];
    v16bf b1 = *(const v16bf*)&Bsf[buf][(wn + 16 + lm) * BK + kh * 16];
    acc[0][0] = __builtin_amdgcn_wmma_f32_16x16x32_bf16(false, a0, false, b0,
                                                        (short)0, acc[0][0], false, false);
    acc[0][1] = __builtin_amdgcn_wmma_f32_16x16x32_bf16(false, a0, false, b1,
                                                        (short)0, acc[0][1], false, false);
    acc[1][0] = __builtin_amdgcn_wmma_f32_16x16x32_bf16(false, a1, false, b0,
                                                        (short)0, acc[1][0], false, false);
    acc[1][1] = __builtin_amdgcn_wmma_f32_16x16x32_bf16(false, a1, false, b1,
                                                        (short)0, acc[1][1], false, false);
  };

  // ---- double-buffered pipeline: one barrier per K-step ----
  gloadA(0); gloadB(0);
  storeA(0); storeB(0);
  __syncthreads();
  int cur = 0;
  for (int k0 = BK; k0 < K; k0 += BK) {
    gloadA(k0); gloadB(k0);     // issue next tile's global_load_b128 early
    compute(cur);               // overlap with WMMA on current tile
    storeA(cur ^ 1); storeB(cur ^ 1);
    __syncthreads();
    cur ^= 1;
  }
  compute(cur);

  // ---- epilogue: C/D layout lane n=L%16; VGPR r -> row m = r + 8*(L/16) ----
#pragma unroll
  for (int mt = 0; mt < 2; ++mt) {
    const int rowb = m0 + wm + mt * 16 + (kh << 3);
#pragma unroll
    for (int nt = 0; nt < 2; ++nt) {
      const int col = n0 + wn + nt * 16 + lm;
      if (col >= N) continue;
#pragma unroll
      for (int r = 0; r < 8; ++r) {
        int row = rowb + r;
        if (row >= M) continue;
        float v = acc[mt][nt][r];
        long long ci = (long long)row * ldc + col;
        if (EPI == 0)      C[ci] = v;
        else if (EPI == 1) C[ci] += gvec[(long long)row * gstride] * v;
        else               C[ci] = (col <= row) ? v * scale : -__builtin_inff();
      }
    }
  }
}

// ---------------------------------------------------------------------------
// LayerNorm: one block per token row of length DDIM
// ---------------------------------------------------------------------------
__global__ void ln_kernel(const float* __restrict__ x, const float* __restrict__ g,
                          const float* __restrict__ b, float* __restrict__ o)
{
  const int n = blockIdx.x;
  const float* xr = x + (long long)n * DDIM;
  float* orow     = o + (long long)n * DDIM;
  __shared__ float red[256];
  const int tid = threadIdx.x;

  float s = 0.f;
  for (int i = tid; i < DDIM; i += 256) s += xr[i];
  red[tid] = s; __syncthreads();
  for (int st = 128; st > 0; st >>= 1) {
    if (tid < st) red[tid] += red[tid + st];
    __syncthreads();
  }
  float mu = red[0] / (float)DDIM;
  __syncthreads();

  float v = 0.f;
  for (int i = tid; i < DDIM; i += 256) { float d = xr[i] - mu; v += d * d; }
  red[tid] = v; __syncthreads();
  for (int st = 128; st > 0; st >>= 1) {
    if (tid < st) red[tid] += red[tid + st];
    __syncthreads();
  }
  float rstd = rsqrtf(red[0] / (float)DDIM + 1e-5f);

  for (int i = tid; i < DDIM; i += 256)
    orow[i] = (xr[i] - mu) * rstd * g[i] + b[i];
}

// ---------------------------------------------------------------------------
// RoPE in-place on contiguous rows of length D. pos = positions[(R/posDiv)%T]
// ---------------------------------------------------------------------------
__global__ void rope_kernel(float* __restrict__ X, const int* __restrict__ positions,
                            int posDiv)
{
  const long long R = blockIdx.x;
  const int t = (int)((R / posDiv) % TT);
  const float p = (float)positions[t];
  float* row = X + R * (long long)DDIM;
  const int e = threadIdx.x;                     // 0..HALF-1
  const float inv = __expf(-((float)e / (float)HALF) * 9.210340372f); // ln(1e4)
  const float f = p * inv;
  const float c = __cosf(f), s = __sinf(f);
  const float a  = row[e];
  const float b2 = row[e + HALF];
  row[e]        = a * c - b2 * s;
  row[e + HALF] = b2 * c + a * s;
}

// ---------------------------------------------------------------------------
// Row softmax over TT columns (scores already hold -inf above the diagonal)
// ---------------------------------------------------------------------------
__global__ void softmax_kernel(float* __restrict__ S)
{
  float* row = S + (long long)blockIdx.x * TT;
  __shared__ float red[256];
  const int tid = threadIdx.x;

  float m = -__builtin_inff();
  for (int i = tid; i < TT; i += 256) m = fmaxf(m, row[i]);
  red[tid] = m; __syncthreads();
  for (int st = 128; st > 0; st >>= 1) {
    if (tid < st) red[tid] = fmaxf(red[tid], red[tid + st]);
    __syncthreads();
  }
  m = red[0]; __syncthreads();

  float s = 0.f;
  for (int i = tid; i < TT; i += 256) { float e = __expf(row[i] - m); row[i] = e; s += e; }
  red[tid] = s; __syncthreads();
  for (int st = 128; st > 0; st >>= 1) {
    if (tid < st) red[tid] += red[tid + st];
    __syncthreads();
  }
  const float inv = 1.f / red[0];
  for (int i = tid; i < TT; i += 256) row[i] *= inv;
}

// Sum per-head attention outputs + residual: x1 = x + sum_h O[n, h*D + d]
__global__ void headsum_kernel(const float* __restrict__ x, const float* __restrict__ O,
                               float* __restrict__ x1)
{
  const int i = blockIdx.x * 256 + threadIdx.x;
  if (i >= NTOK * DDIM) return;
  const int n = i / DDIM, d = i % DDIM;
  float s = x[i];
  const float* orow = O + (long long)n * HDIM + d;
#pragma unroll
  for (int h = 0; h < HH; ++h) s += orow[h * DDIM];
  x1[i] = s;
}

// Router: logits -> softmax(E=8) -> top-2 -> renormalized dense gates (N x E)
__global__ void router_kernel(const float* __restrict__ xn, const float* __restrict__ rw,
                              float* __restrict__ gates)
{
  const int n = blockIdx.x * 256 + threadIdx.x;
  if (n >= NTOK) return;
  const float* xr = xn + (long long)n * DDIM;
  float lg[EE];
  float mx = -__builtin_inff();
#pragma unroll
  for (int e = 0; e < EE; ++e) {
    float s = 0.f;
    const float* w = rw + e * DDIM;
    for (int d = 0; d < DDIM; ++d) s += xr[d] * w[d];
    lg[e] = s; mx = fmaxf(mx, s);
  }
  float sum = 0.f;
#pragma unroll
  for (int e = 0; e < EE; ++e) { lg[e] = __expf(lg[e] - mx); sum += lg[e]; }
#pragma unroll
  for (int e = 0; e < EE; ++e) lg[e] /= sum;
  int i1 = 0;
#pragma unroll
  for (int e = 1; e < EE; ++e) if (lg[e] > lg[i1]) i1 = e;
  int i2 = (i1 == 0) ? 1 : 0;
#pragma unroll
  for (int e = 0; e < EE; ++e) if (e != i1 && lg[e] > lg[i2]) i2 = e;
  const float s2 = lg[i1] + lg[i2];
#pragma unroll
  for (int e = 0; e < EE; ++e) gates[(long long)n * EE + e] = 0.f;
  gates[(long long)n * EE + i1] = lg[i1] / s2;
  gates[(long long)n * EE + i2] = lg[i2] / s2;
}

// h1 = silu(h1) * h3  (in place)
__global__ void silu_mul_kernel(float* __restrict__ h1, const float* __restrict__ h3,
                                long long n)
{
  const long long i = (long long)blockIdx.x * 256 + threadIdx.x;
  if (i >= n) return;
  const float a = h1[i];
  h1[i] = (a / (1.f + __expf(-a))) * h3[i];
}

__global__ void zero_kernel(float* __restrict__ p, long long n)
{
  const long long i = (long long)blockIdx.x * 256 + threadIdx.x;
  if (i < n) p[i] = 0.f;
}

__global__ void add_kernel(const float* __restrict__ a, const float* __restrict__ b,
                           float* __restrict__ o, long long n)
{
  const long long i = (long long)blockIdx.x * 256 + threadIdx.x;
  if (i < n) o[i] = a[i] + b[i];
}

// ---------------------------------------------------------------------------
// Host-side GEMM launcher: only the four used (trB, epi) paths instantiated
// ---------------------------------------------------------------------------
static void launch_gemm(hipStream_t st, const float* A, const float* B, float* C,
                        int M, int N, int K, int lda, int ldb, int ldc, int trB,
                        int Z, int zdiv,
                        long long sA1, long long sA2, long long sB1, long long sB2,
                        long long sC1, long long sC2,
                        int epi, const float* gvec, int gstride, float scale)
{
  dim3 grid(M / BM, N / BN, Z);
#define GO(T, E) gemm_wmma_k<T, E><<<grid, 256, 0, st>>>(A, B, C, M, N, K, lda, ldb, ldc, \
    zdiv, sA1, sA2, sB1, sB2, sC1, sC2, gvec, gstride, scale)
  if (!trB)            GO(0, 0);   // projections, P@V
  else if (epi == 0)   GO(1, 0);   // MoE h1/h3
  else if (epi == 1)   GO(1, 1);   // MoE gated accumulate
  else                 GO(1, 2);   // causal scores
#undef GO
}

extern "C" void kernel_launch(void* const* d_in, const int* in_sizes, int n_in,
                              void* d_out, int out_size, void* d_ws, size_t ws_size,
                              hipStream_t stream)
{
  const float* x      = (const float*)d_in[0];
  const int*   pos    = (const int*)  d_in[1];
  const float* ln1_g  = (const float*)d_in[2];
  const float* ln1_b  = (const float*)d_in[3];
  const float* Mw     = (const float*)d_in[4];   // (D, H*D)
  const float* Vw     = (const float*)d_in[5];   // (D, H*D)
  const float* ln2_g  = (const float*)d_in[6];
  const float* ln2_b  = (const float*)d_in[7];
  const float* rw     = (const float*)d_in[8];   // (E, D)
  const float* w1     = (const float*)d_in[9];   // (E, F, D)
  const float* w2     = (const float*)d_in[10];  // (E, D, F)
  const float* w3     = (const float*)d_in[11];  // (E, F, D)
  float* out          = (float*)d_out;

  // ---- workspace layout (fp32 elements), with MoE overlaid onto dead Q ----
  float* ws  = (float*)d_ws;
  float* xn1 = ws;                              // NTOK*DDIM   (later: K after RoPE)
  float* Q   = xn1 + (long long)NTOK * DDIM;    // NTOK*HDIM
  float* Vv  = Q   + (long long)NTOK * HDIM;    // NTOK*HDIM
  float* S   = Vv  + (long long)NTOK * HDIM;    // BHH*TT*TT
  float* x1  = S   + (long long)BHH * TT * TT;  // NTOK*DDIM
  float* Obuf  = Q;                             // per-head attn out (Q dead after scores)
  float* xn2   = Q;                             // MoE overlays (attn phase done)
  float* gates = xn2 + (long long)NTOK * DDIM;
  float* h1    = gates + (long long)NTOK * EE;
  float* h3    = h1 + (long long)NTOK * FF;
  float* y     = h3 + (long long)NTOK * FF;

  const float scale = 0.044194173824159216f;    // D^-0.5, D=512
  const long long ND = (long long)NTOK * DDIM;

  // ---- 1) LN1 ----
  ln_kernel<<<NTOK, 256, 0, stream>>>(x, ln1_g, ln1_b, xn1);

  // ---- 2) Q = xn1 @ M ; val = xn1 @ V  (output layout [n, h*D+d]) ----
  launch_gemm(stream, xn1, Mw, Q,  NTOK, HDIM, DDIM, DDIM, HDIM, HDIM, 0,
              1, 1, 0,0,0,0,0,0, 0, nullptr, 0, 0.f);
  launch_gemm(stream, xn1, Vw, Vv, NTOK, HDIM, DDIM, DDIM, HDIM, HDIM, 0,
              1, 1, 0,0,0,0,0,0, 0, nullptr, 0, 0.f);

  // ---- 3) RoPE on Q rows (N*H rows), and in-place on xn1 -> K ----
  rope_kernel<<<NTOK * HH, HALF, 0, stream>>>(Q,   pos, HH);
  rope_kernel<<<NTOK,      HALF, 0, stream>>>(xn1, pos, 1);

  // ---- 4) scores[z=b*H+h] = Q_bh @ K_b^T, causal * scale ----
  launch_gemm(stream, Q, xn1, S, TT, TT, DDIM, HDIM, DDIM, TT, 1,
              BHH, HH,
              (long long)TT * HDIM, (long long)DDIM,       // A: b*T*HD + h*D
              (long long)TT * DDIM, 0,                     // B: b*T*D
              (long long)HH * TT * TT, (long long)TT * TT, // C: z*T*T
              2, nullptr, 0, scale);

  // ---- 5) row softmax ----
  softmax_kernel<<<BHH * TT, 256, 0, stream>>>(S);

  // ---- 6) O_bh = P @ val_bh  (into Obuf, [n, h*D+d]) ----
  launch_gemm(stream, S, Vv, Obuf, TT, DDIM, TT, TT, HDIM, HDIM, 0,
              BHH, HH,
              (long long)HH * TT * TT, (long long)TT * TT,
              (long long)TT * HDIM, (long long)DDIM,
              (long long)TT * HDIM, (long long)DDIM,
              0, nullptr, 0, 0.f);

  // ---- 7) head sum + residual ----
  headsum_kernel<<<(int)((ND + 255) / 256), 256, 0, stream>>>(x, Obuf, x1);

  // ---- 8) LN2, router, gates ----
  ln_kernel<<<NTOK, 256, 0, stream>>>(x1, ln2_g, ln2_b, xn2);
  router_kernel<<<NTOK / 256, 256, 0, stream>>>(xn2, rw, gates);

  // ---- 9) MoE: dense per expert, gate-weighted accumulate into y ----
  zero_kernel<<<(int)((ND + 255) / 256), 256, 0, stream>>>(y, ND);
  const long long NF = (long long)NTOK * FF;
  for (int e = 0; e < EE; ++e) {
    const float* w1e = w1 + (long long)e * FF * DDIM;
    const float* w3e = w3 + (long long)e * FF * DDIM;
    const float* w2e = w2 + (long long)e * DDIM * FF;
    // h1 = xn2 @ w1_e^T ; h3 = xn2 @ w3_e^T     (w: F x D, accessed transposed)
    launch_gemm(stream, xn2, w1e, h1, NTOK, FF, DDIM, DDIM, DDIM, FF, 1,
                1, 1, 0,0,0,0,0,0, 0, nullptr, 0, 0.f);
    launch_gemm(stream, xn2, w3e, h3, NTOK, FF, DDIM, DDIM, DDIM, FF, 1,
                1, 1, 0,0,0,0,0,0, 0, nullptr, 0, 0.f);
    silu_mul_kernel<<<(int)((NF + 255) / 256), 256, 0, stream>>>(h1, h3, NF);
    // y += gate[:,e] * (h @ w2_e^T)             (w2_e: D x F, transposed)
    launch_gemm(stream, h1, w2e, y, NTOK, DDIM, FF, FF, FF, DDIM, 1,
                1, 1, 0,0,0,0,0,0, 1, gates + e, EE, 0.f);
  }

  // ---- 10) out = x1 + y ----
  add_kernel<<<(int)((ND + 255) / 256), 256, 0, stream>>>(x1, y, out, ND);
}
